// BPSpikingNet_62723702391151
// MI455X (gfx1250) — matrix operand order repaired and checked by reading, not compiled, not run on hardware
//
#include <hip/hip_runtime.h>

// ---------------------------------------------------------------------------
// CDNA5 (gfx1250) fused GEMM + LIF spiking network.
//   z[t,b,o] = sum_i h[t,b,i] * W[o,i] + b[o]
//   v = 0.5*(v+z); s = (v>=1); v = s?0:v   (v carried in registers over t)
// bf16 WMMA (v_wmma_f32_16x16x32_bf16) with f32 accumulate. A/W tiles are
// staged into double-buffered LDS by the Tensor Data Mover (TDM,
// tensor_load_to_lds + s_wait_tensorcnt), one barrier per K-chunk. TDM's
// pad_interval/pad_amount reproduce the bank-spread LDS row stride (72
// halves = 128B row + 16B pad).
// ---------------------------------------------------------------------------

typedef __bf16 bf16;
typedef __attribute__((ext_vector_type(16))) __bf16 bf16x16;
typedef __attribute__((ext_vector_type(8)))  __bf16 bf16x8;
typedef __attribute__((ext_vector_type(8)))  float  f32x8;

typedef unsigned int       u32;
typedef unsigned long long u64;
typedef __attribute__((ext_vector_type(4))) u32 u32x4;
typedef __attribute__((ext_vector_type(8))) int i32x8;
typedef __attribute__((ext_vector_type(4))) int i32x4;

#define KC          64      // K-chunk per LDS stage (two 16x16x32 WMMA steps)
#define LDS_STRIDE  72      // KC + 8 halves pad (matches TDM pad fields)
#define WG_THREADS  128     // 4 wave32
#define WG_COLS     128     // 4 waves * 2 tiles * 16 columns

// ------------------ fp32 -> zero-padded bf16 pre-pass -----------------------
__global__ __launch_bounds__(256) void cvt_pad_bf16(
    const float* __restrict__ src, bf16* __restrict__ dst,
    int O, int K, int K_pad)
{
    int total = O * K_pad;
    for (int idx = blockIdx.x * blockDim.x + threadIdx.x; idx < total;
         idx += gridDim.x * blockDim.x) {
        int o = idx / K_pad;
        int k = idx - o * K_pad;
        float v = (k < K) ? src[(size_t)o * K + k] : 0.0f;
        dst[idx] = (bf16)v;
    }
}

// zero-pad rows too (for W2: 20 -> 128 rows)
__global__ __launch_bounds__(256) void cvt_pad2_bf16(
    const float* __restrict__ src, bf16* __restrict__ dst,
    int O, int K, int O_pad, int K_pad)
{
    int total = O_pad * K_pad;
    for (int idx = blockIdx.x * blockDim.x + threadIdx.x; idx < total;
         idx += gridDim.x * blockDim.x) {
        int o = idx / K_pad;
        int k = idx - o * K_pad;
        float v = (o < O && k < K) ? src[(size_t)o * K + k] : 0.0f;
        dst[idx] = (bf16)v;
    }
}

// -------------------- TDM 2D tile load: global -> LDS -----------------------
// Loads a (rows x KC) bf16 tile whose rows are `stride_elems` apart into LDS,
// inserting 16B of padding after every 128B row => LDS row stride 144B.
__device__ __forceinline__ void tdm_tile_load(
    u32 lds_off, u64 gaddr, int rows, int tensor_dim0, int tensor_dim1,
    int stride_elems)
{
    u32x4 g0;
    g0.x = 1u;                                   // count=1, user mode
    g0.y = lds_off;                              // lds_addr
    g0.z = (u32)gaddr;                           // global_addr[31:0]
    g0.w = (u32)((gaddr >> 32) & 0x01FFFFFFull)  // global_addr[56:32]
         | (2u << 30);                           // type=2 ("image")
    i32x8 g1;
    g1[0] = (int)((1u << 16)     // data_size = 2 bytes
                | (1u << 20)     // pad_enable
                | (4u << 22)     // pad_interval: 32 DWORDs (=128B) per pad
                | (3u << 25));   // pad_amount: 4 DWORDs (=16B)
    g1[1] = (int)(((u32)tensor_dim0 & 0xFFFFu) << 16);          // dim0[15:0]
    g1[2] = (int)((((u32)tensor_dim0 >> 16) & 0xFFFFu)
                | (((u32)tensor_dim1 & 0xFFFFu) << 16));        // dim0[31:16], dim1[15:0]
    g1[3] = (int)((((u32)tensor_dim1 >> 16) & 0xFFFFu)
                | ((u32)KC << 16));                             // dim1[31:16], tile_dim0=KC
    g1[4] = rows;            // tile_dim1 = rows, tile_dim2 = 0
    g1[5] = stride_elems;    // tensor_dim0_stride[31:0]
    g1[6] = 0;               // stride hi, tensor_dim1_stride (unused for 2D)
    g1[7] = 0;
    i32x4 gz4 = {};
    i32x8 gz8 = {};
    __builtin_amdgcn_tensor_load_to_lds(g0, g1, gz4, gz4, gz8, 0);
}

// --------------------------- fused layer kernel -----------------------------
// A: bf16 [T*B, K_pad] (zero-padded). Wb: bf16 [N_pad, K_pad] (zero-padded).
// Out: spikes, bf16 [T*B, Ostride] or f32 for the final layer.
__global__ __launch_bounds__(WG_THREADS) void lif_gemm_layer(
    const bf16* __restrict__ Ab, const bf16* __restrict__ Wb,
    int K_pad, int N_pad, const float* __restrict__ bias, int N,
    void* __restrict__ Out, int out_is_f32, int Ostride,
    int T, int B)
{
    __shared__ __align__(16) bf16 ldsA[2][16 * LDS_STRIDE];
    __shared__ __align__(16) bf16 ldsW[2][WG_COLS * LDS_STRIDE];

    const int tid     = threadIdx.x;
    const int wave    = tid >> 5;
    const int lane    = tid & 31;
    const int lane_lo = lane & 15;
    const int hi      = lane >> 4;

    const int b0    = blockIdx.y * 16;        // batch rows of this WG
    const int ncol0 = blockIdx.x * WG_COLS;   // output columns of this WG

    // this wave's two 16-wide column tiles
    const int n_g0 = ncol0 + wave * 32 +  0 + lane_lo;
    const int n_g1 = ncol0 + wave * 32 + 16 + lane_lo;

    const float bia0 = (n_g0 < N) ? bias[n_g0] : 0.0f;
    const float bia1 = (n_g1 < N) ? bias[n_g1] : 0.0f;

    float* Of = (float*)Out;
    bf16*  Ob = (bf16*)Out;

    // LDS byte offsets of the double buffers (ptrtoint also marks them escaped)
    const u32 offA0 = (u32)(size_t)&ldsA[0][0];
    const u32 offA1 = (u32)(size_t)&ldsA[1][0];
    const u32 offW0 = (u32)(size_t)&ldsW[0][0];
    const u32 offW1 = (u32)(size_t)&ldsW[1][0];

    // membrane potentials, C/D-matrix register layout: m = hi*8 + r
    f32x8 v0 = {}; f32x8 v1 = {};

    const int nkc = K_pad / KC;
    const u64 Wbase = (u64)(size_t)(Wb + (size_t)ncol0 * K_pad);

    auto issue_tdm = [&](int t, int kc, int pp) {
        // W tile: WG_COLS rows of this WG's columns
        tdm_tile_load(pp ? offW1 : offW0,
                      Wbase + (u64)(kc * KC) * sizeof(bf16),
                      WG_COLS, K_pad, N_pad, K_pad);
        // A tile: 16 batch rows at timestep t
        tdm_tile_load(pp ? offA1 : offA0,
                      (u64)(size_t)(Ab + ((size_t)t * B + b0) * K_pad + kc * KC),
                      16, K_pad, T * B, K_pad);
    };

    // prologue: DMA first chunk into buffer 0
    if (wave == 0) {
        issue_tdm(0, 0, 0);
        __builtin_amdgcn_s_wait_tensorcnt(0);
    }
    int p = 0;

    for (int t = 0; t < T; ++t) {
        f32x8 acc0 = {}; f32x8 acc1 = {};

        for (int kc = 0; kc < nkc; ++kc) {
            __syncthreads();                   // buffer p complete & visible

            const bool last = (t == T - 1) && (kc == nkc - 1);
            if (!last && wave == 0) {
                int t_n = t, kc_n = kc + 1;
                if (kc_n == nkc) { kc_n = 0; ++t_n; }
                issue_tdm(t_n, kc_n, p ^ 1);   // DMA next chunk -> back buffer
            }

            #pragma unroll
            for (int ks = 0; ks < KC; ks += 32) {
                // A fragment (CDNA5 16-bit A layout): lane row m = lane_lo,
                // halves 0..7 = K[ks+hi*8..], halves 8..15 = K[ks+16+hi*8..]
                bf16x16 afrag;
                {
                    bf16x8 alo = *(const bf16x8*)&ldsA[p][lane_lo * LDS_STRIDE + ks + hi * 8];
                    bf16x8 ahi = *(const bf16x8*)&ldsA[p][lane_lo * LDS_STRIDE + ks + 16 + hi * 8];
                    #pragma unroll
                    for (int j = 0; j < 8; ++j) { afrag[j] = alo[j]; afrag[8 + j] = ahi[j]; }
                }
                // B fragment tile 0: lane column n = lane_lo, halves 0..7 = K[ks+hi*8..], halves 8..15 = K[ks+16+hi*8..]
                {
                    const bf16* wrow = &ldsW[p][(wave * 32 + lane_lo) * LDS_STRIDE + ks + hi * 8];
                    bf16x8 blo = *(const bf16x8*)(wrow);
                    bf16x8 bhi = *(const bf16x8*)(wrow + 16);
                    bf16x16 bfrag;
                    #pragma unroll
                    for (int j = 0; j < 8; ++j) { bfrag[j] = blo[j]; bfrag[8 + j] = bhi[j]; }
                    acc0 = __builtin_amdgcn_wmma_f32_16x16x32_bf16(
                        false, afrag, false, bfrag, (short)0, acc0, false, false);
                }
                // B fragment tile 1
                {
                    const bf16* wrow = &ldsW[p][(wave * 32 + 16 + lane_lo) * LDS_STRIDE + ks + hi * 8];
                    bf16x8 blo = *(const bf16x8*)(wrow);
                    bf16x8 bhi = *(const bf16x8*)(wrow + 16);
                    bf16x16 bfrag;
                    #pragma unroll
                    for (int j = 0; j < 8; ++j) { bfrag[j] = blo[j]; bfrag[8 + j] = bhi[j]; }
                    acc1 = __builtin_amdgcn_wmma_f32_16x16x32_bf16(
                        false, afrag, false, bfrag, (short)0, acc1, false, false);
                }
            }

            if (wave == 0) __builtin_amdgcn_s_wait_tensorcnt(0); // DMA done
            p ^= 1;                             // next barrier publishes it
        }

        // ---- LIF epilogue for timestep t ----
        #pragma unroll
        for (int r = 0; r < 8; ++r) {
            const int m = hi * 8 + r;
            const size_t orow = ((size_t)t * B + b0 + m) * (size_t)Ostride;
            {
                float z  = acc0[r] + bia0;
                float vv = 0.5f * (v0[r] + z);
                float s  = (vv >= 1.0f) ? 1.0f : 0.0f;
                v0[r]    = (vv >= 1.0f) ? 0.0f : vv;
                if (n_g0 < N) {
                    if (out_is_f32) Of[orow + n_g0] = s;
                    else            Ob[orow + n_g0] = (bf16)s;
                }
            }
            {
                float z  = acc1[r] + bia1;
                float vv = 0.5f * (v1[r] + z);
                float s  = (vv >= 1.0f) ? 1.0f : 0.0f;
                v1[r]    = (vv >= 1.0f) ? 0.0f : vv;
                if (n_g1 < N) {
                    if (out_is_f32) Of[orow + n_g1] = s;
                    else            Ob[orow + n_g1] = (bf16)s;
                }
            }
        }
    }
}

// ------------------------------- launcher -----------------------------------
extern "C" void kernel_launch(void* const* d_in, const int* in_sizes, int n_in,
                              void* d_out, int out_size, void* d_ws, size_t ws_size,
                              hipStream_t stream) {
    const float* x   = (const float*)d_in[0];
    const float* W0  = (const float*)d_in[1];
    const float* b0v = (const float*)d_in[2];
    const float* W1  = (const float*)d_in[3];
    const float* b1v = (const float*)d_in[4];
    const float* W2  = (const float*)d_in[5];
    const float* b2v = (const float*)d_in[6];
    float* out = (float*)d_out;

    const int T = 100, B = 256, IN = 700, H = 1024, C = 20;
    const int IN_pad = 704;      // K padded to multiple of KC
    const int C_pad  = WG_COLS;  // W2 rows padded to one WG of columns
    const int TB     = T * B;

    // workspace layout (all chunk sizes are multiples of 256 bytes)
    char* ws = (char*)d_ws;
    bf16* W0b = (bf16*)ws;   ws += (size_t)H * IN_pad  * sizeof(bf16);
    bf16* W1b = (bf16*)ws;   ws += (size_t)H * H       * sizeof(bf16);
    bf16* W2b = (bf16*)ws;   ws += (size_t)C_pad * H   * sizeof(bf16);
    bf16* xb  = (bf16*)ws;   ws += (size_t)TB * IN_pad * sizeof(bf16);
    bf16* h0  = (bf16*)ws;   ws += (size_t)TB * H      * sizeof(bf16);
    bf16* h1  = (bf16*)ws;
    (void)ws_size; (void)in_sizes; (void)n_in; (void)out_size;

    // operands -> zero-padded bf16
    cvt_pad_bf16 <<<dim3(1024), dim3(256), 0, stream>>>(W0, W0b, H,  IN, IN_pad);
    cvt_pad_bf16 <<<dim3(1024), dim3(256), 0, stream>>>(W1, W1b, H,  H,  H);
    cvt_pad2_bf16<<<dim3(512),  dim3(256), 0, stream>>>(W2, W2b, C,  H,  C_pad, H);
    cvt_pad_bf16 <<<dim3(8192), dim3(256), 0, stream>>>(x,  xb,  TB, IN, IN_pad);

    const dim3 blk(WG_THREADS);
    // layer 0: xb (bf16, K_pad=704) -> h0 spikes (bf16)
    lif_gemm_layer<<<dim3(H / WG_COLS, B / 16), blk, 0, stream>>>(
        xb, W0b, IN_pad, H,     b0v, H, h0,  0, H, T, B);
    // layer 1: h0 -> h1
    lif_gemm_layer<<<dim3(H / WG_COLS, B / 16), blk, 0, stream>>>(
        h0, W1b, H,      H,     b1v, H, h1,  0, H, T, B);
    // layer 2: h1 -> d_out (f32 spikes, N=20)
    lif_gemm_layer<<<dim3(1, B / 16), blk, 0, stream>>>(
        h1, W2b, H,      C_pad, b2v, C, out, 1, C, T, B);
}
